// GRUWorldModel_34213709480599
// MI455X (gfx1250) — compile-verified, matrix operand order checked by reading, not run on hardware
//
#include <hip/hip_runtime.h>
#include <hip/hip_bf16.h>
#include <stdint.h>

// ---------------------------------------------------------------------------
// GRU world model, MI455X (gfx1250, wave32, WMMA 16x16x32 bf16)
//   rows (B*N = 2048) independent -> 128 persistent workgroups of 16 rows.
//   Hidden state is a bf16 ping-pong pair in LDS (read t&1, write the other),
//   weights pre-swizzled to fragment-major bf16 (coalesced 2x b128 per frag).
// ---------------------------------------------------------------------------

typedef __attribute__((ext_vector_type(16))) __bf16        v16bf;
typedef __attribute__((ext_vector_type(8)))  float         v8f;
typedef __attribute__((ext_vector_type(4)))  unsigned int  u32x4;

#define OBS_D   128
#define ACT_D   32
#define IN_D    160
#define H_D     512
#define BDIM    32
#define TDIM    64
#define NDIM    64
#define HOR     63
#define ROWS    16
#define NTHREADS 512

#define XS_STRIDE  168      // padded LDS row strides -> conflict-free b128 A-frags
#define HBF_STRIDE 520
#define HBUF       (ROWS * HBF_STRIDE)

#define NK_IH   (IN_D / 32)   // 5
#define NK_HH   (H_D / 32)    // 16

#define W_IH_E  (3 * H_D * IN_D)   // 245760
#define W_HH_E  (3 * H_D * H_D)    // 786432
#define W_DEC_E (OBS_D * H_D)      // 65536

__device__ __forceinline__ unsigned short f2bf(float f) {
  unsigned int u = __float_as_uint(f);
  u += 0x7FFFu + ((u >> 16) & 1u);        // round-to-nearest-even
  return (unsigned short)(u >> 16);
}
__device__ __forceinline__ float bf2f(unsigned short b) {
  return __uint_as_float((unsigned int)b << 16);
}

union FragU { u32x4 q[2]; v16bf v; };

// A-fragment from LDS (ISA 7.12.2): lane(<16) K {0..7,16..23}, lane(>=16)
// K {8..15,24..31}; two 16-byte chunks per lane, conflict-free via row padding.
__device__ __forceinline__ v16bf load_fragA(const unsigned short* rowp, int k0, int kh) {
  FragU u;
  u.q[0] = *(const u32x4*)(rowp + k0 + kh * 8);
  u.q[1] = *(const u32x4*)(rowp + k0 + 16 + kh * 8);
  return u.v;
}

// B-fragment from fragment-major packed weights: one contiguous 32B per lane.
__device__ __forceinline__ v16bf load_fragB(const unsigned short* pk, int tile, int nK,
                                            int kt, int lane) {
  const unsigned short* p = pk + (((long)tile * nK + kt) * 32 + lane) * 16;
  FragU u;
  u.q[0] = *(const u32x4*)p;
  u.q[1] = *(const u32x4*)(p + 8);
  return u.v;
}

__device__ __forceinline__ float sigmoidf_(float x) { return 1.0f / (1.0f + __expf(-x)); }
__device__ __forceinline__ float tanhf_(float x) {
  float e = __expf(-2.0f * x);
  return (1.0f - e) / (1.0f + e);
}

#define WMMA_BF16(A, B, C) \
  __builtin_amdgcn_wmma_f32_16x16x32_bf16(false, (A), false, (B), (short)0, (C), false, false)

// ---- weight pre-pack: f32 row-major -> bf16 fragment-major ----------------
// out[((tile*nK + kt)*32 + lane)*16 + e], lane = (kh<<4)|n,
//   e<8 : k = kt*32 + kh*8 + e ;  e>=8 : k = kt*32 + 16 + kh*8 + (e-8)
__global__ void pack_weights(const float* __restrict__ w, unsigned short* __restrict__ out,
                             int K, long total) {
  long i = (long)blockIdx.x * blockDim.x + threadIdx.x;
  if (i >= total) return;
  int e    = (int)(i & 15);
  int lane = (int)((i >> 4) & 31);
  long r   = i >> 9;
  int nK   = K / 32;
  int kt   = (int)(r % nK);
  int tile = (int)(r / nK);
  int n = lane & 15, kh = lane >> 4;
  int k = kt * 32 + ((e < 8) ? (kh * 8 + e) : (16 + kh * 8 + (e - 8)));
  out[i] = f2bf(w[(long)(tile * 16 + n) * K + k]);
}

// ---- fused persistent GRU kernel ------------------------------------------
__global__ void __launch_bounds__(NTHREADS, 1)
gru_wm_kernel(const float* __restrict__ obs, const float* __restrict__ act,
              const unsigned short* __restrict__ wihP,
              const unsigned short* __restrict__ whhP,
              const unsigned short* __restrict__ wdecP,
              const float* __restrict__ bih, const float* __restrict__ bhh,
              const float* __restrict__ bdec,
              float* __restrict__ preds, float* __restrict__ hs) {
  __shared__ __align__(16) unsigned short xs [ROWS * XS_STRIDE];  //  5.25 KB
  __shared__ __align__(16) unsigned short hpp[2 * HBUF];          // 32.5 KB ping-pong h

  const int tid   = threadIdx.x;
  const int wave  = tid >> 5;          // 16 waves
  const int lane  = tid & 31;
  const int laneN = lane & 15;
  const int kh    = lane >> 4;
  const int rowbase = blockIdx.x * ROWS;
  const int c0    = wave * 32;         // wave owns h columns [c0, c0+32)

  for (int i = tid; i < 2 * HBUF; i += NTHREADS) hpp[i] = 0;   // h0 = 0

  float biasR[2], biasZ[2], biasIN[2], biasHN[2];
#pragma unroll
  for (int ct = 0; ct < 2; ++ct) {
    int c = c0 + ct * 16 + laneN;
    biasR[ct]  = bih[c] + bhh[c];
    biasZ[ct]  = bih[512 + c] + bhh[512 + c];
    biasIN[ct] = bih[1024 + c];
    biasHN[ct] = bhh[1024 + c];
  }
  const float biasD = (wave < 8) ? bdec[wave * 16 + laneN] : 0.0f;

  __syncthreads();

#pragma unroll 1
  for (int t = 0; t < HOR; ++t) {
    const unsigned short* hcur = hpp + (t & 1) * HBUF;         // read
    unsigned short*       hnxt = hpp + ((t & 1) ^ 1) * HBUF;   // write

    // ---- stage x_t = [obs | act] as bf16 into LDS -------------------------
    for (int i = tid; i < ROWS * IN_D; i += NTHREADS) {
      int m = i / IN_D, c = i - m * IN_D;
      int r = rowbase + m;
      int bb = r >> 6, nn = r & 63;
      float v = (c < OBS_D)
          ? obs[(((long)bb * TDIM + t) * NDIM + nn) * OBS_D + c]
          : act[(((long)bb * TDIM + t) * NDIM + nn) * ACT_D + (c - OBS_D)];
      xs[m * XS_STRIDE + c] = f2bf(v);
    }
    __syncthreads();   // xs ready; hnxt free (its last readers synced below)

    // ---- two 16-col tiles, strictly sequential (no cross-tile state) ------
#pragma unroll 1
    for (int ct = 0; ct < 2; ++ct) {
      v8f aR, aZ, aIN, aHN;
#pragma unroll
      for (int j = 0; j < 8; ++j) {
        aR[j] = biasR[ct]; aZ[j] = biasZ[ct];
        aIN[j] = biasIN[ct]; aHN[j] = biasHN[ct];
      }
      const int tR = wave * 2 + ct;        // row-tile in gate r block
      const int tZ = 32 + tR;              // +512 rows
      const int tN = 64 + tR;              // +1024 rows

      // x side: K = 160 -> 5 k-steps, accumulate r, z, i_n
      const unsigned short* xrow = xs + laneN * XS_STRIDE;
#pragma unroll 1
      for (int kt = 0; kt < NK_IH; ++kt) {
        v16bf A  = load_fragA(xrow, kt * 32, kh);
        v16bf Br = load_fragB(wihP, tR, NK_IH, kt, lane);
        aR  = WMMA_BF16(A, Br, aR);
        v16bf Bz = load_fragB(wihP, tZ, NK_IH, kt, lane);
        aZ  = WMMA_BF16(A, Bz, aZ);
        v16bf Bn = load_fragB(wihP, tN, NK_IH, kt, lane);
        aIN = WMMA_BF16(A, Bn, aIN);
      }

      // h side: K = 512 -> 16 k-steps, accumulate r, z, h_n
      const unsigned short* hrow = hcur + laneN * HBF_STRIDE;
#pragma unroll 1
      for (int kt = 0; kt < NK_HH; ++kt) {
        v16bf A  = load_fragA(hrow, kt * 32, kh);
        v16bf Br = load_fragB(whhP, tR, NK_HH, kt, lane);
        aR  = WMMA_BF16(A, Br, aR);
        v16bf Bz = load_fragB(whhP, tZ, NK_HH, kt, lane);
        aZ  = WMMA_BF16(A, Bz, aZ);
        v16bf Bn = load_fragB(whhP, tN, NK_HH, kt, lane);
        aHN = WMMA_BF16(A, Bn, aHN);
      }

      // gate fusion; commit straight to the OTHER h buffer (no hazard) ------
      const int col = c0 + ct * 16 + laneN;
#pragma unroll
      for (int j = 0; j < 8; ++j) {
        const int m = kh ? (j + 8) : j;   // C/D layout: lanes 16-31 hold M+8
        float ho = bf2f(hcur[m * HBF_STRIDE + col]);
        float rg = sigmoidf_(aR[j]);
        float zg = sigmoidf_(aZ[j]);
        float ng = tanhf_(aIN[j] + rg * aHN[j]);
        float hn = (1.0f - zg) * ng + zg * ho;
        hnxt[m * HBF_STRIDE + col] = f2bf(hn);
        int rr = rowbase + m;
        int bb = rr >> 6, nn = rr & 63;
        hs[(((long)bb * HOR + t) * NDIM + nn) * H_D + col] = hn;
      }
    }
    __syncthreads();   // h_new fully committed in hnxt

    // ---- decoder: preds = h_new @ w_dec^T + b_dec (waves 0..7) ------------
    if (wave < 8) {
      const int pc = wave * 16 + laneN;
      v8f ap;
#pragma unroll
      for (int j = 0; j < 8; ++j) ap[j] = biasD;
      const unsigned short* hrow = hnxt + laneN * HBF_STRIDE;
#pragma unroll 1
      for (int kt = 0; kt < NK_HH; ++kt) {
        v16bf A  = load_fragA(hrow, kt * 32, kh);
        v16bf Bd = load_fragB(wdecP, wave, NK_HH, kt, lane);
        ap = WMMA_BF16(A, Bd, ap);
      }
#pragma unroll
      for (int j = 0; j < 8; ++j) {
        const int m = kh ? (j + 8) : j;
        int rr = rowbase + m;
        int bb = rr >> 6, nn = rr & 63;
        preds[(((long)bb * HOR + t) * NDIM + nn) * OBS_D + pc] = ap[j];
      }
    }
    // no barrier needed here: next step reads hnxt (decoder also read hnxt),
    // and next step's writes go to hcur, whose readers sync at the next
    // __syncthreads() after xs staging. xs rewrite is fenced by that sync too.
    __syncthreads();
  }
}

// ---------------------------------------------------------------------------
extern "C" void kernel_launch(void* const* d_in, const int* in_sizes, int n_in,
                              void* d_out, int out_size, void* d_ws, size_t ws_size,
                              hipStream_t stream) {
  (void)in_sizes; (void)n_in; (void)out_size; (void)ws_size;

  const float* obs  = (const float*)d_in[0];
  const float* act  = (const float*)d_in[1];
  const float* wih  = (const float*)d_in[2];
  const float* whh  = (const float*)d_in[3];
  const float* bih  = (const float*)d_in[4];
  const float* bhh  = (const float*)d_in[5];
  const float* wdec = (const float*)d_in[6];
  const float* bdec = (const float*)d_in[7];
  // d_in[8] = horizon (device scalar); statically 63 for this shape.

  float* preds = (float*)d_out;
  float* hs    = preds + (long)BDIM * HOR * NDIM * OBS_D;

  unsigned short* wihP  = (unsigned short*)d_ws;       // fragment-major bf16
  unsigned short* whhP  = wihP + W_IH_E;
  unsigned short* wdecP = whhP + W_HH_E;

  pack_weights<<<(W_IH_E  + 255) / 256, 256, 0, stream>>>(wih,  wihP,  IN_D, (long)W_IH_E);
  pack_weights<<<(W_HH_E  + 255) / 256, 256, 0, stream>>>(whh,  whhP,  H_D,  (long)W_HH_E);
  pack_weights<<<(W_DEC_E + 255) / 256, 256, 0, stream>>>(wdec, wdecP, H_D,  (long)W_DEC_E);

  gru_wm_kernel<<<(BDIM * NDIM) / ROWS, NTHREADS, 0, stream>>>(
      obs, act, wihP, whhP, wdecP, bih, bhh, bdec, preds, hs);
}
